// KNNtorch_90409061580965
// MI455X (gfx1250) — matrix operand order repaired and checked by reading, not compiled, not run on hardware
//
#include <hip/hip_runtime.h>
#include <hip/hip_bf16.h>

// ---------------------------------------------------------------------------
// kNN cosine-sim classifier for MI455X (gfx1250, wave32, WMMA)
//   1) normalize queries -> f16
//   2) per-row inverse L2 norms of features
//   3) WMMA f16 GEMM  sims[q][c] = (qn . feat_c) * inv_norm[c]  (f32 acc)
//      - A tile via GLOBAL_LOAD_ASYNC_TO_LDS_B128 (ASYNCcnt)
//      - branch-free inner loop (clamped loads, guarded stores)
//      - B tile: batch 4 global_load_b128 before converting (MLP)
//   4) per-query top-32 select + exp(sim/T) class histogram
// ---------------------------------------------------------------------------

typedef __attribute__((ext_vector_type(16))) _Float16 v16h;
typedef __attribute__((ext_vector_type(8)))  _Float16 v8h;
typedef __attribute__((ext_vector_type(4)))  _Float16 v4h;
typedef __attribute__((ext_vector_type(8)))  float    v8f;
typedef __attribute__((ext_vector_type(4)))  float    f32x4;
typedef __attribute__((ext_vector_type(4)))  int      i32x4;

#if defined(__has_builtin)
#  if __has_builtin(__builtin_amdgcn_global_load_async_to_lds_b128) && \
      __has_builtin(__builtin_amdgcn_s_wait_asynccnt)
#    define USE_ASYNC_LDS 1
#  endif
#endif
#ifndef USE_ASYNC_LDS
#  define USE_ASYNC_LDS 0
#endif

// builtin expects (int4 AS1*, int4 AS3*, imm offset, imm cpol)
typedef __attribute__((address_space(1))) i32x4* as1_i4;
typedef __attribute__((address_space(3))) i32x4* as3_i4;

static constexpr int QDIM    = 512;    // feature dimension
static constexpr int NCLS    = 1000;   // classes
static constexpr int TOPK    = 32;
static constexpr float TEMP_INV = 1.0f / 0.07f;

static constexpr int BM = 128;   // queries per block
static constexpr int BN = 128;   // candidates per block
static constexpr int BK = 32;    // K step (matches 16x16x32 WMMA)
static constexpr int LDT = BK + 8; // padded LDS row stride (halves)

__device__ inline int imin(int a, int b) { return a < b ? a : b; }

__device__ inline float waveReduceSum(float v) {
#pragma unroll
  for (int off = 16; off >= 1; off >>= 1)
    v += __shfl_xor(v, off, 32);
  return v;
}

// -------------------- kernel 1: normalize queries to f16 -------------------
__global__ __launch_bounds__(256)
void knn_norm_queries(const float* __restrict__ q, _Float16* __restrict__ qn, int B) {
  const int wid  = threadIdx.x >> 5;
  const int lane = threadIdx.x & 31;
  const int row  = blockIdx.x * 8 + wid;           // one wave per row
  if (row >= B) return;
  const float* rp = q + (size_t)row * QDIM;
  f32x4 vals[4];
  float ss = 0.f;
#pragma unroll
  for (int j = 0; j < 4; ++j) {
    vals[j] = *(const f32x4*)(rp + 4 * (lane + 32 * j));
#pragma unroll
    for (int t = 0; t < 4; ++t) ss += vals[j][t] * vals[j][t];
  }
  ss = waveReduceSum(ss);
  const float inv = 1.0f / fmaxf(sqrtf(ss), 1e-12f);
  _Float16* op = qn + (size_t)row * QDIM;
#pragma unroll
  for (int j = 0; j < 4; ++j) {
    v4h h;
#pragma unroll
    for (int t = 0; t < 4; ++t) h[t] = (_Float16)(vals[j][t] * inv);
    *(v4h*)(op + 4 * (lane + 32 * j)) = h;
  }
}

// -------------------- kernel 2: feature inverse norms ----------------------
__global__ __launch_bounds__(256)
void knn_feat_norms(const float* __restrict__ f, float* __restrict__ inv, int N) {
  const int wid  = threadIdx.x >> 5;
  const int lane = threadIdx.x & 31;
  const int row  = blockIdx.x * 8 + wid;           // one wave per row
  if (row >= N) return;
  const float* rp = f + (size_t)row * QDIM;
  float ss = 0.f;
#pragma unroll
  for (int j = 0; j < 4; ++j) {
    f32x4 v = *(const f32x4*)(rp + 4 * (lane + 32 * j));
#pragma unroll
    for (int t = 0; t < 4; ++t) ss += v[t] * v[t];
  }
  ss = waveReduceSum(ss);
  if (lane == 0) inv[row] = 1.0f / fmaxf(sqrtf(ss), 1e-12f);
}

// -------------------- kernel 3: WMMA GEMM ----------------------------------
// 256 threads = 8 waves in a 2(M) x 4(N) grid; wave region 64x32 = 8 tiles.
// Loads are clamped (never OOB), only the final stores are guarded: WMMA
// rows/columns are independent, so clamped duplicate rows are harmless.
__global__ __launch_bounds__(256)
void knn_gemm(const _Float16* __restrict__ qn, const float* __restrict__ feat,
              const float* __restrict__ inv, float* __restrict__ sims,
              int B, int N) {
  __shared__ _Float16 aT[BM * LDT];   // queries tile, row-major [q][k]
  __shared__ _Float16 bT[BN * LDT];   // feature tile, row-major [cand][k]

  const int tid   = threadIdx.x;
  const int lane  = tid & 31;
  const int w     = tid >> 5;
  const int wm    = w >> 2;       // 0..1
  const int wn    = w & 3;        // 0..3
  const int m0    = blockIdx.y * BM;
  const int n0    = blockIdx.x * BN;
  const int lhalf = lane & 15;
  const bool hiL  = lane >= 16;

  v8f acc[4][2];
  const v8f vz = {0.f,0.f,0.f,0.f,0.f,0.f,0.f,0.f};
#pragma unroll
  for (int m = 0; m < 4; ++m)
#pragma unroll
    for (int n = 0; n < 2; ++n) acc[m][n] = vz;

  for (int k0 = 0; k0 < QDIM; k0 += BK) {
    // ---- A tile: 128 x 32 halves = 512 x v8h; 2 async chunks per thread ----
#if USE_ASYNC_LDS
#pragma unroll
    for (int j = 0; j < 2; ++j) {
      const int idx = tid + j * 256;      // 0..511
      const int r   = idx >> 2;           // 4 v8h per row
      const int c   = idx & 3;
      const int gq  = imin(m0 + r, B - 1);
      __builtin_amdgcn_global_load_async_to_lds_b128(
          (as1_i4)(qn + (size_t)gq * QDIM + k0 + c * 8),
          (as3_i4)&aT[r * LDT + c * 8], 0, 0);
    }
#else
#pragma unroll
    for (int j = 0; j < 2; ++j) {
      const int idx = tid + j * 256;
      const int r   = idx >> 2;
      const int c   = idx & 3;
      const int gq  = imin(m0 + r, B - 1);
      v8h av = *(const v8h*)(qn + (size_t)gq * QDIM + k0 + c * 8);
      *(v8h*)&aT[r * LDT + c * 8] = av;
    }
#endif
    // ---- B tile: 128 x 32 f32 -> f16 ----
    // Each thread owns 2 chunks of 8 contiguous floats. Issue all 4
    // global_load_b128 first (distinct regs => overlapped), then convert
    // and store one b128 per chunk to LDS.
    f32x4 fv[2][2];
#pragma unroll
    for (int j = 0; j < 2; ++j) {
      const int idx = tid + j * 256;        // 0..511 (chunks of 8 floats)
      const int r   = idx >> 2;             // 4 chunks per 32-float row
      const int c   = idx & 3;
      const int gc  = imin(n0 + r, N - 1);
      const float* p = feat + (size_t)gc * QDIM + k0 + c * 8;
      fv[j][0] = *(const f32x4*)(p);
      fv[j][1] = *(const f32x4*)(p + 4);
    }
#pragma unroll
    for (int j = 0; j < 2; ++j) {
      const int idx = tid + j * 256;
      const int r   = idx >> 2;
      const int c   = idx & 3;
      v8h hv;
#pragma unroll
      for (int t = 0; t < 4; ++t) {
        hv[t]     = (_Float16)fv[j][0][t];
        hv[4 + t] = (_Float16)fv[j][1][t];
      }
      *(v8h*)&bT[r * LDT + c * 8] = hv;
    }
    // prefetch next K chunk of the feature tile (global_prefetch_b8)
    if (k0 + BK < QDIM) {
      const int r = tid >> 2, c = tid & 3;
      const int gc = imin(n0 + (r & (BN - 1)), N - 1);
      __builtin_prefetch(feat + (size_t)gc * QDIM + k0 + BK + c * 8, 0, 1);
    }
#if USE_ASYNC_LDS
    __builtin_amdgcn_s_wait_asynccnt(0);
#endif
    __syncthreads();

    // ---- fragments per ISA lane layout ----
    v16h afr[4];
#pragma unroll
    for (int m = 0; m < 4; ++m) {
      const int row = wm * 64 + m * 16 + lhalf;
      const int kb  = hiL ? 8 : 0;        // lanes>=16 hold K 8-15 / 24-31
      v8h lo = *(const v8h*)&aT[row * LDT + kb];
      v8h hi = *(const v8h*)&aT[row * LDT + kb + 16];
#pragma unroll
      for (int t = 0; t < 8; ++t) { afr[m][t] = lo[t]; afr[m][8 + t] = hi[t]; }
    }
    v16h bfr[2];
#pragma unroll
    for (int n = 0; n < 2; ++n) {
      const int row = wn * 32 + n * 16 + lhalf;   // candidate = B column
      const int kb  = hiL ? 8 : 0;                // lanes>=16 hold K 8-15 / 24-31
      v8h lo = *(const v8h*)&bT[row * LDT + kb];
      v8h hi = *(const v8h*)&bT[row * LDT + kb + 16];
#pragma unroll
      for (int t = 0; t < 8; ++t) { bfr[n][t] = lo[t]; bfr[n][8 + t] = hi[t]; }
    }
#pragma unroll
    for (int m = 0; m < 4; ++m)
#pragma unroll
      for (int n = 0; n < 2; ++n)
        acc[m][n] = __builtin_amdgcn_wmma_f32_16x16x32_f16(
            false, afr[m], false, bfr[n], (short)0, acc[m][n], false, false);
    __syncthreads();
  }

  // ---- epilogue: scale by candidate inverse norm, store sims ----
#pragma unroll
  for (int n = 0; n < 2; ++n) {
    const int c = n0 + wn * 32 + n * 16 + lhalf;   // N = lane%16
    const bool ok = c < N;
    const float iv = inv[imin(c, N - 1)];
#pragma unroll
    for (int m = 0; m < 4; ++m) {
      const int qb = m0 + wm * 64 + m * 16 + (hiL ? 8 : 0);
#pragma unroll
      for (int r = 0; r < 8; ++r) {
        const int qrow = qb + r;                   // M = r (+8 for hi lanes)
        if (ok && qrow < B)
          sims[(size_t)qrow * N + c] = acc[m][n][r] * iv;
      }
    }
  }
}

// -------------------- kernel 4: top-32 select + class histogram ------------
static constexpr int TTH = 128;   // threads per query block

__global__ __launch_bounds__(TTH)
void knn_topk(const float* __restrict__ sims, const int* __restrict__ labels,
              float* __restrict__ out, int N) {
  __shared__ unsigned long long cand[TTH * TOPK];  // 32 KB
  __shared__ unsigned long long redK[TTH];
  __shared__ int redS[TTH];
  __shared__ float cls[NCLS];

  const int tid = threadIdx.x;
  const int qi  = blockIdx.x;
  const float* srow = sims + (size_t)qi * N;

  // per-thread unordered top-32 over strided slice (keys: sortable-f32|idx)
  unsigned long long best[TOPK];
#pragma unroll
  for (int j = 0; j < TOPK; ++j) best[j] = 0ull;
  unsigned long long curMin = 0ull;
  int minPos = 0;

  for (int i = tid; i < N; i += TTH) {
    unsigned u = __float_as_uint(srow[i]);
    u ^= (u >> 31) ? 0xFFFFFFFFu : 0x80000000u;    // monotone f32 -> u32
    const unsigned long long key = ((unsigned long long)u << 32) | (unsigned)i;
    if (key > curMin) {
#pragma unroll
      for (int j = 0; j < TOPK; ++j)
        if (j == minPos) best[j] = key;            // predicated reg write
      curMin = best[0]; minPos = 0;
#pragma unroll
      for (int j = 1; j < TOPK; ++j)
        if (best[j] < curMin) { curMin = best[j]; minPos = j; }
    }
  }
#pragma unroll
  for (int j = 0; j < TOPK; ++j) cand[tid * TOPK + j] = best[j];
  for (int c = tid; c < NCLS; c += TTH) cls[c] = 0.f;
  __syncthreads();

  // 32 rounds of block-wide argmax tournament; order-invariant accumulation
  for (int round = 0; round < TOPK; ++round) {
    unsigned long long mk = 0ull;
    int ms = tid * TOPK;
#pragma unroll
    for (int j = 0; j < TOPK; ++j) {
      const unsigned long long k = cand[tid * TOPK + j];
      if (k > mk) { mk = k; ms = tid * TOPK + j; }
    }
    redK[tid] = mk; redS[tid] = ms;
    __syncthreads();
    for (int s = TTH / 2; s > 0; s >>= 1) {
      if (tid < s && redK[tid + s] > redK[tid]) {
        redK[tid] = redK[tid + s]; redS[tid] = redS[tid + s];
      }
      __syncthreads();
    }
    if (tid == 0) {
      const unsigned long long k = redK[0];
      unsigned u = (unsigned)(k >> 32);
      u = (u >> 31) ? (u ^ 0x80000000u) : ~u;      // inverse transform
      const float v = __uint_as_float(u);
      const int idx = (int)(k & 0xFFFFFFFFull);
      const int lab = labels[idx];
      if (lab >= 0 && lab < NCLS) cls[lab] += __expf(v * TEMP_INV);
      cand[redS[0]] = 0ull;                        // invalidate winner
    }
    __syncthreads();
  }
  for (int c = tid; c < NCLS; c += TTH)
    out[(size_t)qi * NCLS + c] = cls[c];
}

// ---------------------------------------------------------------------------
extern "C" void kernel_launch(void* const* d_in, const int* in_sizes, int n_in,
                              void* d_out, int out_size, void* d_ws, size_t ws_size,
                              hipStream_t stream) {
  const float* queries = (const float*)d_in[0];
  const float* feat    = (const float*)d_in[1];
  const int*   labels  = (const int*)d_in[2];
  float*       out     = (float*)d_out;

  const int B = in_sizes[0] / QDIM;   // 256
  const int N = in_sizes[2];          // 500000

  // workspace layout: [qn f16][inv_norm f32][sims f32]
  char* ws = (char*)d_ws;
  _Float16* qn = (_Float16*)ws;
  size_t off = (((size_t)B * QDIM * sizeof(_Float16)) + 255) & ~(size_t)255;
  float* invn = (float*)(ws + off);
  off += (((size_t)N * sizeof(float)) + 255) & ~(size_t)255;
  float* sims = (float*)(ws + off);

  knn_norm_queries<<<dim3((B + 7) / 8), 256, 0, stream>>>(queries, qn, B);
  knn_feat_norms<<<dim3((N + 7) / 8), 256, 0, stream>>>(feat, invn, N);
  dim3 gg((N + BN - 1) / BN, (B + BM - 1) / BM);
  knn_gemm<<<gg, 256, 0, stream>>>(qn, feat, invn, sims, B, N);
  knn_topk<<<dim3(B), TTH, 0, stream>>>(sims, labels, out, N);
}
